// CombinedGraphLayer_30288109371695
// MI455X (gfx1250) — matrix operand order repaired and checked by reading, not compiled, not run on hardware
//
#include <hip/hip_runtime.h>

// ---------------------------------------------------------------------------
// CDNA5 (gfx1250) implementation of CombinedGraphLayer (LSH-binned GHConv).
// All GEMMs run through v_wmma_f32_16x16x32_bf16 (bf16 in, fp32 accum).
// K4 activation staging uses GLOBAL_LOAD_ASYNC_TO_LDS_B128 (ASYNCcnt).
// ---------------------------------------------------------------------------

#define BATCH   16
#define NPTS    12800
#define FDIM    256
#define DDIM    128
#define NBINS   100
#define PBIN    128
#define NROT    50

typedef __attribute__((ext_vector_type(8)))  float  v8f;
typedef __attribute__((ext_vector_type(8)))  __bf16 bf16x8;
typedef __attribute__((ext_vector_type(16))) __bf16 bf16x16;

// ---------------------------------------------------------------------------
// WMMA helpers
// ---------------------------------------------------------------------------
__device__ __forceinline__ v8f wmma_bf16(bf16x16 a, bf16x16 b, v8f c) {
  // D = A(16x32) * B(32x16) + C, fp32 accumulate
  return __builtin_amdgcn_wmma_f32_16x16x32_bf16(
      /*neg_a=*/false, a, /*neg_b=*/false, b,
      /*c_mod=*/(short)0, c, /*reuse_a=*/false, /*reuse_b=*/false);
}

// A fragment (16x32, bf16) from row-major [M][K] storage (LDS or global).
// lane m = lane&15, khalf = lane>>4; elems e -> K = kt*32+khalf*16+e.
// Two 16-byte loads.
__device__ __forceinline__ bf16x16 load_afrag(const __bf16* base, int ldk,
                                              int m0, int kt, int lane) {
  int m = m0 + (lane & 15);
  int kh = lane >> 4;
  const __bf16* p = base + (size_t)m * ldk + kt * 32 + kh * 16;
  bf16x8 lo = *(const bf16x8*)p;
  bf16x8 hi = *(const bf16x8*)(p + 8);
  return __builtin_shufflevector(lo, hi, 0, 1, 2, 3, 4, 5, 6, 7,
                                 8, 9, 10, 11, 12, 13, 14, 15);
}

// B fragment (32x16, bf16) where memory is laid out "N-major": element (k,n)
// at base[n*ldk + k].  Lane n = lane&15, khalf = lane>>4, elems e -> K =
// kt*32 + khalf*16 + e  (16 contiguous bf16 = one 32-byte load).
__device__ __forceinline__ bf16x16 load_bfrag_nmajor(const __bf16* base, int ldk,
                                                     int n0, int kt, int lane) {
  int n = n0 + (lane & 15);
  int kh = lane >> 4;
  const __bf16* p = base + (size_t)n * ldk + kt * 32 + kh * 16;
  return *(const bf16x16*)p;
}

// B fragment from pre-packed weights: fragment (kt,nt) is 512 contiguous
// bf16, lane-major (16 per lane).
__device__ __forceinline__ bf16x16 load_bfrag_packed(const __bf16* wp, int ntiles,
                                                     int kt, int nt, int lane) {
  return *(const bf16x16*)(wp + (((size_t)(kt * ntiles + nt)) << 9) + (lane << 4));
}

__device__ __forceinline__ float elu1f(float v) {
  return v > 0.0f ? v : (expf(v) - 1.0f);
}

// ---------------------------------------------------------------------------
// K0: pack a row-major fp32 (K x N) weight into fragment-ready bf16 layout.
// One 32-thread block per (kt, nt) fragment.
// ---------------------------------------------------------------------------
__global__ void pack_b_kernel(const float* __restrict__ src,
                              __bf16* __restrict__ dst, int K, int Nn) {
  int ntiles = Nn >> 4;
  int frag = blockIdx.x;
  int kt = frag / ntiles;
  int nt = frag - kt * ntiles;
  int lane = threadIdx.x & 31;
  int n = (nt << 4) + (lane & 15);
  int kb = (kt << 5) + ((lane >> 4) << 4);
  __bf16* out = dst + (((size_t)frag) << 9) + (lane << 4);
#pragma unroll
  for (int e = 0; e < 16; ++e)
    out[e] = (__bf16)src[(size_t)(kb + e) * Nn + n];
  (void)K;
}

// ---------------------------------------------------------------------------
// K1: x_dist = elu(x@w_d1 + b_d1) @ w_d2 + b_d2, then LSH bin index.
// One block = 128 rows, 8 waves, dynamic LDS: 128KB.
//   region0 [0,64K)  : x tile bf16 (128x256)  -> later x_dist fp32 (128x128)
//   region1 [64K,128K): h tile bf16 (128x256) -> later rot cols fp32 (50x128)
// ---------------------------------------------------------------------------
__global__ __launch_bounds__(256) void k1_ffn_kernel(
    const float* __restrict__ x, const int* __restrict__ msk,
    const float* __restrict__ rot, const float* __restrict__ b_d1,
    const float* __restrict__ b_d2, const __bf16* __restrict__ wpd1,
    const __bf16* __restrict__ wpd2, float* __restrict__ xdist,
    int* __restrict__ binidx) {
  extern __shared__ char smem[];
  __bf16* xA = (__bf16*)smem;              // 128 x 256 bf16
  float* xd = (float*)smem;                // 128 x 128 fp32 (reuse)
  __bf16* hB = (__bf16*)(smem + 65536);    // 128 x 256 bf16
  float* rotl = (float*)(smem + 65536);    // 50 x 128 fp32 (reuse)

  const int tid = threadIdx.x;
  const int lane = tid & 31;
  const int wid = tid >> 5;
  const int m0 = wid << 4;
  const int b = blockIdx.x / NBINS;
  const int row0 = (blockIdx.x - b * NBINS) * PBIN;

  // ---- stage x tile (fp32 -> bf16) ----
  {
    const float* xg = x + ((size_t)b * NPTS + row0) * FDIM;
    for (int i = tid; i < (PBIN * FDIM) / 4; i += 256) {
      float4 v = *(const float4*)(xg + (size_t)i * 4);
      int o = i * 4;
      xA[o + 0] = (__bf16)v.x;
      xA[o + 1] = (__bf16)v.y;
      xA[o + 2] = (__bf16)v.z;
      xA[o + 3] = (__bf16)v.w;
    }
  }
  __syncthreads();

  // ---- GEMM1: h = elu(x @ w_d1 + b_d1) ----
  for (int nt = 0; nt < 16; ++nt) {
    v8f acc = {};
#pragma unroll
    for (int kt = 0; kt < 8; ++kt) {
      bf16x16 a = load_afrag(xA, FDIM, m0, kt, lane);
      bf16x16 bb = load_bfrag_packed(wpd1, 16, kt, nt, lane);
      acc = wmma_bf16(a, bb, acc);
    }
    int col = (nt << 4) + (lane & 15);
    float bias = b_d1[col];
#pragma unroll
    for (int g = 0; g < 8; ++g) {
      int r = m0 + ((lane >> 4) << 3) + g;
      hB[r * FDIM + col] = (__bf16)elu1f(acc[g] + bias);
    }
  }
  __syncthreads();

  // ---- GEMM2: xd = h @ w_d2 + b_d2 ----
  for (int nt = 0; nt < 8; ++nt) {
    v8f acc = {};
#pragma unroll
    for (int kt = 0; kt < 8; ++kt) {
      bf16x16 a = load_afrag(hB, FDIM, m0, kt, lane);
      bf16x16 bb = load_bfrag_packed(wpd2, 8, kt, nt, lane);
      acc = wmma_bf16(a, bb, acc);
    }
    int col = (nt << 4) + (lane & 15);
    float bias = b_d2[col];
#pragma unroll
    for (int g = 0; g < 8; ++g) {
      int r = m0 + ((lane >> 4) << 3) + g;
      float v = acc[g] + bias;
      xd[r * DDIM + col] = v;
      xdist[((size_t)b * NPTS + row0 + r) * DDIM + col] = v;
    }
  }
  __syncthreads();

  // ---- stage rotation columns: rotl[r][k] = rot[k][r], r<50 ----
  for (int i = tid; i < NROT * DDIM; i += 256) {
    int rr = i >> 7;
    int k = i & 127;
    rotl[i] = rot[(size_t)k * 100 + rr];
  }
  __syncthreads();

  // ---- LSH binning: argmax over [mul, -mul] (first-occurrence ties) ----
  if (tid < PBIN) {
    const float* xr = xd + tid * DDIM;
    float best1 = -3.0e38f, best2 = -3.0e38f;
    int i1 = 0, i2 = 0;
    for (int r = 0; r < NROT; ++r) {
      const float* rp = rotl + r * DDIM;
      float s = 0.0f;
#pragma unroll 4
      for (int k = 0; k < DDIM; ++k) s += xr[k] * rp[k];
      if (s > best1) { best1 = s; i1 = r; }
      if (-s > best2) { best2 = -s; i2 = r; }
    }
    int bid = (best1 >= best2) ? i1 : (NROT + i2);
    int row = row0 + tid;
    if (msk[(size_t)b * NPTS + row] == 0) bid += NBINS - 1;
    binidx[(size_t)b * NPTS + row] = bid;
  }
}

// ---------------------------------------------------------------------------
// K2: stable counting sort of bin indices (values < 256) per batch.
// ---------------------------------------------------------------------------
__global__ __launch_bounds__(256) void k2_sort_kernel(
    const int* __restrict__ binidx, int* __restrict__ bins) {
  __shared__ int cnt[256];
  __shared__ int offs[256];
  int b = blockIdx.x;
  int t = threadIdx.x;
  const int* bi = binidx + (size_t)b * NPTS;
  int c = 0;
  for (int i = 0; i < NPTS; ++i) c += (bi[i] == t) ? 1 : 0;
  cnt[t] = c;
  __syncthreads();
  if (t == 0) {
    int s = 0;
    for (int v = 0; v < 256; ++v) { offs[v] = s; s += cnt[v]; }
  }
  __syncthreads();
  int o = offs[t];
  int* out = bins + (size_t)b * NPTS;
  for (int i = 0; i < NPTS; ++i)
    if (bi[i] == t) out[o++] = i;
}

// ---------------------------------------------------------------------------
// K3: per-bin gather + Gaussian adjacency (G = xm @ xm^T via WMMA) + norms.
// One block per (batch, bin).  Static LDS ~35 KB.
// ---------------------------------------------------------------------------
__global__ __launch_bounds__(256) void k3_adj_kernel(
    const float* __restrict__ x, const float* __restrict__ xdist,
    const int* __restrict__ msk, const int* __restrict__ bins,
    __bf16* __restrict__ xb0, __bf16* __restrict__ dm,
    float* __restrict__ normw, float* __restrict__ mskbw) {
  __shared__ __bf16 xm[PBIN * DDIM];  // 32 KB
  __shared__ float na[PBIN];
  __shared__ float mk[PBIN];
  __shared__ float indeg[PBIN];
  __shared__ int bidx[PBIN];

  const int bg = blockIdx.x;
  const int b = bg / NBINS;
  const int bin = bg - b * NBINS;
  const int tid = threadIdx.x;
  const int lane = tid & 31;
  const int wid = tid >> 5;
  const int m0 = wid << 4;

  if (tid < PBIN) {
    int orig = bins[(size_t)b * NPTS + bin * PBIN + tid];
    bidx[tid] = orig;
    float m = (msk[(size_t)b * NPTS + orig] != 0) ? 1.0f : 0.0f;
    mk[tid] = m;
    mskbw[(size_t)bg * PBIN + tid] = m;
    na[tid] = 0.0f;
    indeg[tid] = 0.0f;
  }
  __syncthreads();

  // gather masked x_dist -> xm (bf16) + row norms, and binned x -> xb0
  {
    int p = tid >> 1;
    int half = tid & 1;
    float m = mk[p];
    const float* src = xdist + ((size_t)b * NPTS + bidx[p]) * DDIM + half * 64;
    float s = 0.0f;
#pragma unroll 4
    for (int k = 0; k < 64; ++k) {
      float v = src[k] * m;
      s += v * v;
      xm[p * DDIM + half * 64 + k] = (__bf16)v;
    }
    atomicAdd(&na[p], s);

    const float* xs = x + ((size_t)b * NPTS + bidx[p]) * FDIM + half * 128;
    __bf16* xo = xb0 + ((size_t)bg * PBIN + p) * FDIM + half * 128;
    for (int k = 0; k < 128; k += 4) {
      float4 v = *(const float4*)(xs + k);
      xo[k + 0] = (__bf16)v.x;
      xo[k + 1] = (__bf16)v.y;
      xo[k + 2] = (__bf16)v.z;
      xo[k + 3] = (__bf16)v.w;
    }
  }
  __syncthreads();

  // G = xm @ xm^T  (A row-major; B "n-major" contiguous from same buffer)
  __bf16* dmg = dm + (size_t)bg * PBIN * PBIN;
  for (int nt = 0; nt < 8; ++nt) {
    v8f acc = {};
#pragma unroll
    for (int kt = 0; kt < 4; ++kt) {
      bf16x16 a = load_afrag(xm, DDIM, m0, kt, lane);
      bf16x16 bb = load_bfrag_nmajor(xm, DDIM, nt << 4, kt, lane);
      acc = wmma_bf16(a, bb, acc);
    }
    int j = (nt << 4) + (lane & 15);
#pragma unroll
    for (int g = 0; g < 8; ++g) {
      int i = m0 + ((lane >> 4) << 3) + g;
      float d2 = na[i] + na[j] - 2.0f * acc[g];
      float d = sqrtf(fmaxf(d2, 1e-6f));
      float a = expf(-0.1f * d);
      a = fminf(fmaxf(a, 0.0f), 1.0f) * mk[i] * mk[j];
      atomicAdd(&indeg[i], a);
      dmg[(size_t)i * PBIN + j] = (__bf16)a;
    }
  }
  __syncthreads();

  if (tid < PBIN) {
    float dg = fminf(fmaxf(indeg[tid], 0.0f), 1000.0f);
    normw[(size_t)bg * PBIN + tid] = rsqrtf(dg + 1e-6f) * mk[tid];
  }
}

// ---------------------------------------------------------------------------
// K4: one GHConv layer per bin.  Dynamic LDS ~130 KB:
//   xin (128x256 bf16, 64K) | tT (256x128 bf16, 64K) | norm/mk/bins
// Input activations staged via GLOBAL_LOAD_ASYNC_TO_LDS_B128 (ASYNCcnt).
// ---------------------------------------------------------------------------
template <bool LAST>
__global__ __launch_bounds__(256) void k4_ghconv_kernel(
    const __bf16* __restrict__ xin_g, const __bf16* __restrict__ dm,
    const float* __restrict__ normw, const float* __restrict__ mskbw,
    const int* __restrict__ bins, const __bf16* __restrict__ wp_th,
    const __bf16* __restrict__ wp_wh, const __bf16* __restrict__ wp_wt,
    const float* __restrict__ bt, __bf16* __restrict__ out_b,
    float* __restrict__ out_f) {
  extern __shared__ char smem4[];
  __bf16* xin = (__bf16*)smem4;                   // 128 x 256
  __bf16* tT = (__bf16*)(smem4 + 65536);          // 256 x 128 (feature-major)
  float* nrm = (float*)(smem4 + 131072);          // 128
  float* mk = nrm + PBIN;                         // 128
  int* bidx = (int*)(mk + PBIN);                  // 128

  const int bg = blockIdx.x;
  const int b = bg / NBINS;
  const int bin = bg - b * NBINS;
  const int tid = threadIdx.x;
  const int lane = tid & 31;
  const int wid = tid >> 5;
  const int m0 = wid << 4;

  // ---- async stage: 64KB of binned activations, global -> LDS ----
  // CDNA5 GLOBAL_LOAD_ASYNC_TO_LDS_B128 (GV mode): VDST VGPR = LDS byte
  // address (low 32 bits of the generic LDS pointer), VADDR = 64-bit global
  // address.  Tracked with ASYNCcnt; drained with s_wait_asynccnt before the
  // workgroup barrier.
  {
    const char* src = (const char*)(xin_g + (size_t)bg * PBIN * FDIM);
    unsigned lds_base = (unsigned)(uintptr_t)xin;
#pragma unroll
    for (int it = 0; it < 16; ++it) {
      unsigned off = (unsigned)(it * 256 + tid) * 16u;  // 4096 x 16B chunks
      unsigned lds_addr = lds_base + off;
      unsigned long long gaddr = (unsigned long long)(uintptr_t)(src + off);
      asm volatile("global_load_async_to_lds_b128 %0, %1, off"
                   :
                   : "v"(lds_addr), "v"(gaddr)
                   : "memory");
    }
  }
  if (tid < PBIN) {
    nrm[tid] = normw[(size_t)bg * PBIN + tid];
    mk[tid] = mskbw[(size_t)bg * PBIN + tid];
    if (LAST) bidx[tid] = bins[(size_t)b * NPTS + bin * PBIN + tid];
  }
  asm volatile("s_wait_asynccnt 0x0" ::: "memory");
  __syncthreads();

  const float mrow = mk[m0 + (lane & 15)];
  const bf16x16 zfrag = {};

  // ---- Phase A: t = (x*msk @ theta) * norm, stored transposed in LDS ----
  for (int nt = 0; nt < 16; ++nt) {
    __builtin_prefetch((const void*)(wp_th + (((size_t)(nt + 1)) << 9)), 0, 1);
    v8f acc = {};
#pragma unroll
    for (int kt = 0; kt < 8; ++kt) {
      bf16x16 a = load_afrag(xin, FDIM, m0, kt, lane);
      bf16x16 am = (mrow != 0.0f) ? a : zfrag;
      bf16x16 bb = load_bfrag_packed(wp_th, 16, kt, nt, lane);
      acc = wmma_bf16(am, bb, acc);
    }
    int col = (nt << 4) + (lane & 15);
#pragma unroll
    for (int g = 0; g < 8; ++g) {
      int row = m0 + ((lane >> 4) << 3) + g;
      tT[col * PBIN + row] = (__bf16)(acc[g] * nrm[row]);
    }
  }
  __syncthreads();

  // ---- Phase B: f_hom = (adj @ t) * norm ; f_het ; gate ; combine ----
  const __bf16* adj = dm + (size_t)bg * PBIN * PBIN;
  for (int nt = 0; nt < 16; ++nt) {
    v8f ah = {};
#pragma unroll
    for (int kt = 0; kt < 4; ++kt) {
      bf16x16 a = load_afrag(adj, PBIN, m0, kt, lane);
      bf16x16 bb = load_bfrag_nmajor(tT, PBIN, nt << 4, kt, lane);
      ah = wmma_bf16(a, bb, ah);
    }
    v8f ae = {};
    v8f ag = {};
#pragma unroll
    for (int kt = 0; kt < 8; ++kt) {
      bf16x16 a = load_afrag(xin, FDIM, m0, kt, lane);
      bf16x16 am = (mrow != 0.0f) ? a : zfrag;
      bf16x16 bh = load_bfrag_packed(wp_wh, 16, kt, nt, lane);
      ae = wmma_bf16(am, bh, ae);
      bf16x16 bg16 = load_bfrag_packed(wp_wt, 16, kt, nt, lane);
      ag = wmma_bf16(a, bg16, ag);
    }
    int col = (nt << 4) + (lane & 15);
    float bias = bt[col];
#pragma unroll
    for (int g = 0; g < 8; ++g) {
      int row = m0 + ((lane >> 4) << 3) + g;
      float hom = ah[g] * nrm[row];
      float gate = 1.0f / (1.0f + expf(-(ag[g] + bias)));
      float o = gate * hom + (1.0f - gate) * ae[g];
      o = elu1f(o) * mk[row];
      if (LAST)
        out_f[((size_t)b * NPTS + bidx[row]) * FDIM + col] = o;
      else
        out_b[((size_t)bg * PBIN + row) * FDIM + col] = (__bf16)o;
    }
  }
}

// ---------------------------------------------------------------------------
// Host launcher
// ---------------------------------------------------------------------------
extern "C" void kernel_launch(void* const* d_in, const int* in_sizes, int n_in,
                              void* d_out, int out_size, void* d_ws, size_t ws_size,
                              hipStream_t stream) {
  (void)in_sizes; (void)n_in; (void)out_size; (void)ws_size;

  const float* x    = (const float*)d_in[0];
  const int*   msk  = (const int*)d_in[1];
  const float* rot  = (const float*)d_in[2];
  const float* w_d1 = (const float*)d_in[3];
  const float* b_d1 = (const float*)d_in[4];
  const float* w_d2 = (const float*)d_in[5];
  const float* b_d2 = (const float*)d_in[6];
  const float* wt0  = (const float*)d_in[7];
  const float* bt0  = (const float*)d_in[8];
  const float* wh0  = (const float*)d_in[9];
  const float* th0  = (const float*)d_in[10];
  const float* wt1  = (const float*)d_in[11];
  const float* bt1  = (const float*)d_in[12];
  const float* wh1  = (const float*)d_in[13];
  const float* th1  = (const float*)d_in[14];
  float* out = (float*)d_out;

  char* ws = (char*)d_ws;
  // workspace layout (bytes)
  __bf16* wpd1 = (__bf16*)(ws + 0);          // 131072
  __bf16* wpd2 = (__bf16*)(ws + 131072);     // 65536
  __bf16* wth0 = (__bf16*)(ws + 196608);     // 131072
  __bf16* wwh0 = (__bf16*)(ws + 327680);
  __bf16* wwt0 = (__bf16*)(ws + 458752);
  __bf16* wth1 = (__bf16*)(ws + 589824);
  __bf16* wwh1 = (__bf16*)(ws + 720896);
  __bf16* wwt1 = (__bf16*)(ws + 851968);
  float*  xdist = (float*)(ws + 983040);                 // B*N*DD*4 = 104857600
  __bf16* xb1   = (__bf16*)(ws + 983040);                // reuse (same bytes)
  __bf16* xb0   = (__bf16*)(ws + 105840640);             // 104857600
  __bf16* dmw   = (__bf16*)(ws + 210698240);             // 52428800
  float*  normw = (float*)(ws + 263127040);              // 819200
  float*  mskbw = (float*)(ws + 263946240);              // 819200
  int*    binidx = (int*)(ws + 264765440);               // 819200
  int*    bins   = (int*)(ws + 265584640);               // 819200

  const int LDS_K1 = 131072;
  const int LDS_K4 = 131072 + 3 * 512;
  hipFuncSetAttribute(reinterpret_cast<const void*>(k1_ffn_kernel),
                      hipFuncAttributeMaxDynamicSharedMemorySize, LDS_K1);
  hipFuncSetAttribute(reinterpret_cast<const void*>(k4_ghconv_kernel<false>),
                      hipFuncAttributeMaxDynamicSharedMemorySize, LDS_K4);
  hipFuncSetAttribute(reinterpret_cast<const void*>(k4_ghconv_kernel<true>),
                      hipFuncAttributeMaxDynamicSharedMemorySize, LDS_K4);

  // ---- K0: pack weights into WMMA B-fragment layout (bf16) ----
  pack_b_kernel<<<128, 32, 0, stream>>>(w_d1, wpd1, 256, 256);
  pack_b_kernel<<<64, 32, 0, stream>>>(w_d2, wpd2, 256, 128);
  pack_b_kernel<<<128, 32, 0, stream>>>(th0, wth0, 256, 256);
  pack_b_kernel<<<128, 32, 0, stream>>>(wh0, wwh0, 256, 256);
  pack_b_kernel<<<128, 32, 0, stream>>>(wt0, wwt0, 256, 256);
  pack_b_kernel<<<128, 32, 0, stream>>>(th1, wth1, 256, 256);
  pack_b_kernel<<<128, 32, 0, stream>>>(wh1, wwh1, 256, 256);
  pack_b_kernel<<<128, 32, 0, stream>>>(wt1, wwt1, 256, 256);

  // ---- K1: ffn_dist + LSH bin index ----
  k1_ffn_kernel<<<BATCH * NBINS, 256, LDS_K1, stream>>>(
      x, msk, rot, b_d1, b_d2, wpd1, wpd2, xdist, binidx);

  // ---- K2: stable counting sort -> bins_split ----
  k2_sort_kernel<<<BATCH, 256, 0, stream>>>(binidx, bins);

  // ---- K3: gather + Gaussian adjacency + norms ----
  k3_adj_kernel<<<BATCH * NBINS, 256, 0, stream>>>(
      x, xdist, msk, bins, xb0, dmw, normw, mskbw);

  // ---- K4: two GHConv layers; layer 1 scatters fp32 to d_out ----
  k4_ghconv_kernel<false><<<BATCH * NBINS, 256, LDS_K4, stream>>>(
      xb0, dmw, normw, mskbw, bins, wth0, wwh0, wwt0, bt0, xb1, nullptr);
  k4_ghconv_kernel<true><<<BATCH * NBINS, 256, LDS_K4, stream>>>(
      xb1, dmw, normw, mskbw, bins, wth1, wwh1, wwt1, bt1, nullptr, out);
}
